// NCA3D_15109694948179
// MI455X (gfx1250) — matrix operand order repaired and checked by reading, not compiled, not run on hardware
//
#include <hip/hip_runtime.h>
#include <hip/hip_bf16.h>

// ---------------------------------------------------------------------------
// 3D Neural CA step, fused for MI455X (gfx1250, wave32, WMMA).
//   Pass A: perception stencil (async-LDS double-buffered halo) ->
//           bf16 WMMA GEMM (64->96 relu, 96->16) -> masked update to ws.
//   Pass B: pre/post life masks (3x3x3 maxpool) -> final output.
// ---------------------------------------------------------------------------

typedef __attribute__((ext_vector_type(16))) __bf16    v16bf;
typedef __attribute__((ext_vector_type(8)))  float     v8f;
typedef __attribute__((ext_vector_type(4)))  unsigned  u32x4;

union BF16x16 { v16bf v; u32x4 q[2]; };

#define GDIM 64
#define G3   (64 * 64 * 64)
#define CCH  16
#define HID  96
#define TX   8
#define TY   8
#define TZ   4

// Padded LDS strides (elements). 72*2=144B (36 words) and 104*2=208B (52
// words) are 16B-aligned for ds_load_b128 and generate 16 distinct banks
// across 16 consecutive rows (gcd(36,64)=gcd(52,64)=4, cycle 16).
#define PSTR  72   // percep rows (K=64 + pad)
#define W1STR 72   // w1 rows
#define HSTR  104  // hid rows (K=96 + pad)
#define W2STR 104  // w2 rows

// LDS carve (bytes, every segment 16B aligned):
#define OFF_W2S   (96 * W1STR * 2)                    // 13824
#define OFF_B1S   (OFF_W2S + 16 * W2STR * 2)          // 17152
#define OFF_CHAN  (OFF_B1S + 96 * 4)                  // 17536
#define OFF_DBUF  (OFF_CHAN + 2 * 600 * 4)            // 22336 (double buffer)
#define OFF_PERC  (OFF_DBUF + 256 * 17 * 4)           // 39744
#define OFF_HID   (OFF_PERC + 256 * PSTR * 2)         // 76608
#define SMEM_A    (OFF_HID + 256 * HSTR * 2)          // 129856 bytes

#if defined(__has_builtin)
#if __has_builtin(__builtin_amdgcn_global_load_async_to_lds_b32) && \
    __has_builtin(__builtin_amdgcn_s_wait_asynccnt)
#define USE_ASYNC 1
#endif
#endif
#ifndef USE_ASYNC
#define USE_ASYNC 0
#endif

#define AS1 __attribute__((address_space(1)))
#define AS3 __attribute__((address_space(3)))

__device__ __forceinline__ int iclamp(int v, int lo, int hi) {
  return v < lo ? lo : (v > hi ? hi : v);
}

__device__ __forceinline__ unsigned short f2bf(float f) {
  unsigned u = __builtin_bit_cast(unsigned, f);
  u += 0x7FFFu + ((u >> 16) & 1u);                    // round-to-nearest-even
  return (unsigned short)(u >> 16);
}

#if USE_ASYNC
// Per-lane global->LDS async dword copy (ASYNCcnt-tracked).
// Builtin params are (AS1 int*, AS3 int*, imm offset, imm cpol).
// Generic->AS3: flat LDS addresses map to LDS via addr[31:0] (ISA 10.2).
__device__ __forceinline__ void async_copy_b32(const float* gsrc, float* ldst) {
  __builtin_amdgcn_global_load_async_to_lds_b32(
      (AS1 int*)(unsigned long long)(const void*)gsrc,
      (AS3 int*)(unsigned int)(unsigned long long)(const void*)ldst, 0, 0);
}
#endif

// ---------------------------------------------------------------------------
// Pass A
// ---------------------------------------------------------------------------
__global__ __launch_bounds__(256) void nca_fused(
    const float* __restrict__ state, const float* __restrict__ rand_u,
    const float* __restrict__ w1, const float* __restrict__ b1,
    const float* __restrict__ w2, float* __restrict__ ns) {
  extern __shared__ char smem[];
  unsigned short* w1s    = (unsigned short*)smem;
  unsigned short* w2s    = (unsigned short*)(smem + OFF_W2S);
  float*          b1s    = (float*)(smem + OFF_B1S);
  float*          chan   = (float*)(smem + OFF_CHAN);   // 2 x 600
  float*          dbuf   = (float*)(smem + OFF_DBUF);
  unsigned short* percep = (unsigned short*)(smem + OFF_PERC);
  unsigned short* hidb   = (unsigned short*)(smem + OFF_HID);

  const int tid  = threadIdx.x;
  const int lane = tid & 31;
  const int wave = tid >> 5;
  const int lx = tid & 7, ly = (tid >> 3) & 7, lz = tid >> 6;
  const int x0 = blockIdx.x * TX, y0 = blockIdx.y * TY;
  const int zt = blockIdx.z & 15, nb = blockIdx.z >> 4;
  const int z0 = zt * TZ;
  const size_t nbase = (size_t)nb * CCH * G3;

  // ---- stage weights into LDS (bf16) -------------------------------------
  for (int i = tid; i < HID * 64; i += 256) {
    int r = i >> 6, k = i & 63;
    w1s[r * W1STR + k] = f2bf(w1[i]);
  }
  for (int i = tid; i < CCH * HID; i += 256) {
    int r = i / 96, k = i - r * 96;
    w2s[r * W2STR + k] = f2bf(w2[i]);
  }
  if (tid < HID) b1s[tid] = b1[tid];

  // ---- halo loader: 10x10x6 edge-clamped tile of one channel -------------
  auto loadChan = [&](int c, float* dst) {
    const float* sc = state + nbase + (size_t)c * G3;
    for (int i = tid; i < 600; i += 256) {
      int ix = i % 10, t2 = i / 10;
      int iy = t2 % 10, iz = t2 / 10;
      int gx = iclamp(x0 - 1 + ix, 0, GDIM - 1);
      int gy = iclamp(y0 - 1 + iy, 0, GDIM - 1);
      int gz = iclamp(z0 - 1 + iz, 0, GDIM - 1);
#if USE_ASYNC
      async_copy_b32(sc + gz * 4096 + gy * 64 + gx, dst + i);
#else
      dst[i] = sc[gz * 4096 + gy * 64 + gx];
#endif
    }
  };

  // ---- perception: per-channel stencil, async double-buffered halos ------
  loadChan(0, chan);
  for (int c = 0; c < CCH; ++c) {
    if (c + 1 < CCH) loadChan(c + 1, chan + 600 * ((c + 1) & 1));
#if USE_ASYNC
    // In-order completion: cnt <= (this wave's prefetch issues) => channel c
    // has fully landed while channel c+1 stays in flight behind the compute.
    if (c + 1 < CCH) {
      if (wave <= 2) __builtin_amdgcn_s_wait_asynccnt(3);
      else           __builtin_amdgcn_s_wait_asynccnt(2);
    } else {
      __builtin_amdgcn_s_wait_asynccnt(0);
    }
#endif
    __syncthreads();
    const float* cb = chan + 600 * (c & 1);

    float A_[3], B_[3], Cs[3];
#pragma unroll
    for (int dz = 0; dz < 3; ++dz) {
      float av[3], bv[3];
#pragma unroll
      for (int dy = 0; dy < 3; ++dy) {
        const float* row = cb + (lz + dz) * 100 + (ly + dy) * 10 + lx;
        float v0 = row[0], v1 = row[1], v2 = row[2];
        av[dy] = v0 + 2.f * v1 + v2;   // g along w
        bv[dy] = v0 - v2;              // s along w
      }
      A_[dz] = av[0] + 2.f * av[1] + av[2];  // g_h g_w
      B_[dz] = av[0] - av[2];                // s_h g_w
      Cs[dz] = bv[0] + 2.f * bv[1] + bv[2];  // g_h s_w
    }
    float ident = cb[(lz + 1) * 100 + (ly + 1) * 10 + lx + 1];
    float sx = (A_[0] - A_[2]) * 0.0625f;                 // s_d g_h g_w
    float sy = (B_[0] + 2.f * B_[1] + B_[2]) * 0.0625f;   // g_d s_h g_w
    float sz = (Cs[0] + 2.f * Cs[1] + Cs[2]) * 0.0625f;   // g_d g_h s_w

    unsigned short* pr = percep + tid * PSTR;   // feature x = f*16 + c
    pr[c]      = f2bf(ident);
    pr[16 + c] = f2bf(sx);
    pr[32 + c] = f2bf(sy);
    pr[48 + c] = f2bf(sz);
    __syncthreads();
  }

  // ---- WMMA MLP: each wave owns 32 voxels = two 16-row A tiles -----------
  const int laneHi = (lane >> 4) & 1;
  const int lml    = lane & 15;
  const int m0     = wave * 32;

  // A fragments (perception): per lane two contiguous 8xbf16 runs -> b128.
  BF16x16 a1[2][2];  // [tile][kChunk]
#pragma unroll
  for (int t = 0; t < 2; ++t) {
    const unsigned short* ap = percep + (m0 + t * 16 + lml) * PSTR;
#pragma unroll
    for (int ch = 0; ch < 2; ++ch) {
      a1[t][ch].q[0] = *(const u32x4*)(ap + ch * 32 + laneHi * 16);
      a1[t][ch].q[1] = *(const u32x4*)(ap + ch * 32 + laneHi * 16 + 8);
    }
  }

  // GEMM1: hid[32x96] = percep[32x64] @ W1^T + b1, relu (B shared over t)
#pragma unroll
  for (int nt = 0; nt < 6; ++nt) {
    const int nb16 = nt * 16;
    const float bias = b1s[nb16 + lml];
    v8f acc[2];
#pragma unroll
    for (int t = 0; t < 2; ++t)
#pragma unroll
      for (int r = 0; r < 8; ++r) acc[t][r] = bias;
#pragma unroll
    for (int ch = 0; ch < 2; ++ch) {
      BF16x16 bfr;
      const unsigned short* bp =
          w1s + (nb16 + lml) * W1STR + ch * 32 + laneHi * 16;
      bfr.q[0] = *(const u32x4*)(bp);
      bfr.q[1] = *(const u32x4*)(bp + 8);
#pragma unroll
      for (int t = 0; t < 2; ++t)
        acc[t] = __builtin_amdgcn_wmma_f32_16x16x32_bf16(
            false, a1[t][ch].v, false, bfr.v, (short)0, acc[t], false, false);
    }
#pragma unroll
    for (int t = 0; t < 2; ++t) {
      unsigned short* hq = hidb + nb16 + lml;
#pragma unroll
      for (int r = 0; r < 8; ++r) {
        float hv = acc[t][r] > 0.f ? acc[t][r] : 0.f;
        hq[(m0 + t * 16 + r + laneHi * 8) * HSTR] = f2bf(hv);
      }
    }
  }

  // GEMM2: delta[32x16] = hid[32x96] @ W2^T (K=96 -> 3 chunks, B shared)
  {
    BF16x16 a2[2][3];
#pragma unroll
    for (int t = 0; t < 2; ++t) {
      const unsigned short* hp = hidb + (m0 + t * 16 + lml) * HSTR;
#pragma unroll
      for (int ch = 0; ch < 3; ++ch) {
        a2[t][ch].q[0] = *(const u32x4*)(hp + ch * 32 + laneHi * 16);
        a2[t][ch].q[1] = *(const u32x4*)(hp + ch * 32 + laneHi * 16 + 8);
      }
    }
    v8f acc2[2];
#pragma unroll
    for (int t = 0; t < 2; ++t)
#pragma unroll
      for (int r = 0; r < 8; ++r) acc2[t][r] = 0.f;
#pragma unroll
    for (int ch = 0; ch < 3; ++ch) {
      BF16x16 b2;
      const unsigned short* bp2 = w2s + lml * W2STR + ch * 32 + laneHi * 16;
      b2.q[0] = *(const u32x4*)(bp2);
      b2.q[1] = *(const u32x4*)(bp2 + 8);
#pragma unroll
      for (int t = 0; t < 2; ++t)
        acc2[t] = __builtin_amdgcn_wmma_f32_16x16x32_bf16(
            false, a2[t][ch].v, false, b2.v, (short)0, acc2[t], false, false);
    }
    // D layout: lane = channel N, VGPR r -> voxel M; transpose via LDS.
#pragma unroll
    for (int t = 0; t < 2; ++t) {
      float* dq = dbuf + lml;
#pragma unroll
      for (int r = 0; r < 8; ++r)
        dq[(m0 + t * 16 + r + laneHi * 8) * 17] = acc2[t][r];
    }
  }
  __syncthreads();

  // ---- masked state update, coalesced channel-major writeback ------------
  const int gx = x0 + lx, gy = y0 + ly, gz = z0 + lz;
  const int fl = gz * 4096 + gy * 64 + gx;
  const float ru  = rand_u[(size_t)nb * G3 + fl];
  const float msk = (ru < 0.5f) ? 1.f : 0.f;
#pragma unroll
  for (int c = 0; c < CCH; ++c) {
    size_t idx = nbase + (size_t)c * G3 + fl;
    ns[idx] = state[idx] + dbuf[tid * 17 + c] * msk;
  }
}

// ---------------------------------------------------------------------------
// Pass B: life masks (pre from old alpha, post from new alpha) and output
// ---------------------------------------------------------------------------
__global__ __launch_bounds__(256) void nca_mask(
    const float* __restrict__ state, const float* __restrict__ ns,
    float* __restrict__ out) {
  __shared__ float oldA[600];
  __shared__ float newA[600];

  const int tid = threadIdx.x;
  const int lx = tid & 7, ly = (tid >> 3) & 7, lz = tid >> 6;
  const int x0 = blockIdx.x * TX, y0 = blockIdx.y * TY;
  const int zt = blockIdx.z & 15, nb = blockIdx.z >> 4;
  const int z0 = zt * TZ;
  const size_t nbase = (size_t)nb * CCH * G3;

  const float* so = state + nbase + (size_t)3 * G3;  // old alpha
  const float* sn = ns + nbase + (size_t)3 * G3;     // new alpha
  for (int i = tid; i < 600; i += 256) {
    int ix = i % 10, t2 = i / 10;
    int iy = t2 % 10, iz = t2 / 10;
    int gx = iclamp(x0 - 1 + ix, 0, GDIM - 1);
    int gy = iclamp(y0 - 1 + iy, 0, GDIM - 1);
    int gz = iclamp(z0 - 1 + iz, 0, GDIM - 1);
    int g = gz * 4096 + gy * 64 + gx;
#if USE_ASYNC
    async_copy_b32(so + g, oldA + i);
    async_copy_b32(sn + g, newA + i);
#else
    oldA[i] = so[g];
    newA[i] = sn[g];
#endif
  }
#if USE_ASYNC
  __builtin_amdgcn_s_wait_asynccnt(0);
#endif
  __syncthreads();

  float mo = -3.4e38f, mn = -3.4e38f;
#pragma unroll
  for (int dz = 0; dz < 3; ++dz)
#pragma unroll
    for (int dy = 0; dy < 3; ++dy)
#pragma unroll
      for (int dx = 0; dx < 3; ++dx) {
        int idx = (lz + dz) * 100 + (ly + dy) * 10 + (lx + dx);
        mo = fmaxf(mo, oldA[idx]);
        mn = fmaxf(mn, newA[idx]);
      }
  const float mask = (mo > 0.1f && mn > 0.1f) ? 1.f : 0.f;

  const int fl = (z0 + lz) * 4096 + (y0 + ly) * 64 + (x0 + lx);
#pragma unroll
  for (int c = 0; c < CCH; ++c) {
    size_t idx = nbase + (size_t)c * G3 + fl;
    out[idx] = ns[idx] * mask;
  }
}

// ---------------------------------------------------------------------------
extern "C" void kernel_launch(void* const* d_in, const int* in_sizes, int n_in,
                              void* d_out, int out_size, void* d_ws,
                              size_t ws_size, hipStream_t stream) {
  const float* state  = (const float*)d_in[0];
  const float* rand_u = (const float*)d_in[1];
  const float* w1     = (const float*)d_in[2];
  const float* b1     = (const float*)d_in[3];
  const float* w2     = (const float*)d_in[4];
  float* out = (float*)d_out;
  float* ns  = (float*)d_ws;  // new_state scratch: 4*16*64^3*4 = 64 MB

  dim3 grid(GDIM / TX, GDIM / TY, (GDIM / TZ) * 4);  // (8, 8, 16*N)
  dim3 block(256);
  nca_fused<<<grid, block, SMEM_A, stream>>>(state, rand_u, w1, b1, w2, ns);
  nca_mask<<<grid, block, 0, stream>>>(state, ns, out);
}
